// GraphAttentionLayerV3_2_18949395710597
// MI455X (gfx1250) — matrix operand order repaired and checked by reading, not compiled, not run on hardware
//
#include <hip/hip_runtime.h>

// Fused GAT layer for MI455X (gfx1250, wave32, WMMA).
// One workgroup (256 threads = 8 waves) per batch element b.
//  - g = h[b] @ W_l^T via v_wmma_f32_16x16x32_f16, operands streamed straight
//    from L2-resident global memory (no staging), g kept in LDS (f32).
//  - adj is staged to LDS with GLOBAL_LOAD_ASYNC_TO_LDS (ASYNCcnt), fully
//    overlapped with the GEMM; drained with s_wait_asynccnt before use.
//  - per head: e via VALU (LeakyReLU inside the contraction -> not a matmul),
//    e2 = e @ adj^T via WMMA, row softmax, attn = a @ g_head via WMMA -> d_out.
// No HBM intermediates (d_ws unused).

typedef __attribute__((ext_vector_type(16))) _Float16 v16h;
typedef __attribute__((ext_vector_type(8)))  float    v8f;

// Async-to-LDS builtin takes (addrspace(1) int*, addrspace(3) int*, imm, imm)
// (clang prints addrspace(1) as '__device__ int *' in HIP mode).
typedef __attribute__((address_space(1))) int  ga_int;
typedef __attribute__((address_space(3))) int  ls_int;

#define WMMA_F16(A, B, C) \
  __builtin_amdgcn_wmma_f32_16x16x32_f16(false, (A), false, (B), (short)0, (C), false, false)

#if defined(__has_builtin)
# if __has_builtin(__builtin_amdgcn_global_load_async_to_lds_b32)
#  define HAVE_ASYNC_LDS 1
# else
#  define HAVE_ASYNC_LDS 0
# endif
# if __has_builtin(__builtin_amdgcn_s_wait_asynccnt)
#  define HAVE_WAIT_ASYNC_BUILTIN 1
# else
#  define HAVE_WAIT_ASYNC_BUILTIN 0
# endif
#else
# define HAVE_ASYNC_LDS 0
# define HAVE_WAIT_ASYNC_BUILTIN 0
#endif

enum : int {
  N_NODES   = 57,
  F_IN      = 128,
  N_HEADS   = 8,
  F_HID     = 32,
  F_OUT     = 256,   // N_HEADS * F_HID
  NP        = 64,    // N padded to multiple of 16
  GS_STRIDE = 257,   // g row stride (floats); odd -> conflict-free columns
  ES_STRIDE = 65     // adj/e/e2 row stride (floats)
};

// --- WMMA fragment loaders (contiguous k order, same in A and B, wave32) --------------------
// A (16x32, f16): lane L -> row m = L%16 ; element e -> k = (L/16)*16 + e

// A-fragment straight from global h[b] (f32), rows >= 57 masked to zero
// (address clamped in-bounds, value multiplied by 0).
__device__ __forceinline__ v16h a_frag_global(const float* __restrict__ hb,
                                              int mbase, int kbase, int lane) {
  v16h a;
  const int m = mbase + (lane & 15);
  const int mc = (m < N_NODES) ? m : (N_NODES - 1);
  const float zmask = (m < N_NODES) ? 1.0f : 0.0f;
  const float* p = hb + (size_t)mc * F_IN + kbase + ((lane >> 4) << 4);
#pragma unroll
  for (int e = 0; e < 16; ++e) a[e] = (_Float16)(p[e] * zmask);
  return a;
}
// A-fragment from f32 LDS (e_s / e2_s)
__device__ __forceinline__ v16h a_frag_f32(const float* s, int mbase, int stride,
                                           int kbase, int lane) {
  v16h a;
  const float* p = s + (size_t)(mbase + (lane & 15)) * stride + kbase + ((lane >> 4) << 4);
#pragma unroll
  for (int e = 0; e < 16; ++e) a[e] = (_Float16)p[e];
  return a;
}
// B (32x16, f16): lane L -> col n = L%16 ; element e -> k = (L/16)*16 + e
// "T" variant: source stored as [n][k] (i.e. B = src^T)
__device__ __forceinline__ v16h b_fragT_global(const float* __restrict__ w,
                                               int nbase, int kbase, int lane) {
  v16h b;
  const float* p = w + (size_t)(nbase + (lane & 15)) * F_IN + kbase + ((lane >> 4) << 4);
#pragma unroll
  for (int e = 0; e < 16; ++e) b[e] = (_Float16)p[e];  // 64B contiguous per lane
  return b;
}
__device__ __forceinline__ v16h b_fragT_f32(const float* s, int nbase, int stride,
                                            int kbase, int lane) {
  v16h b;
  const float* p = s + (size_t)(nbase + (lane & 15)) * stride + kbase + ((lane >> 4) << 4);
#pragma unroll
  for (int e = 0; e < 16; ++e) b[e] = (_Float16)p[e];
  return b;
}
// "N" variant: source stored as [k][n]
__device__ __forceinline__ v16h b_fragN_f32(const float* s, int kbase, int stride,
                                            int nbase, int lane) {
  v16h b;
  const float* p = s + (size_t)(kbase + ((lane >> 4) << 4)) * stride + nbase + (lane & 15);
#pragma unroll
  for (int e = 0; e < 16; ++e) b[e] = (_Float16)p[(size_t)e * stride];
  return b;
}
// C/D (16x16, f32, 8 VGPRs): VGPR r, lane L -> (m = r + 8*(L>=16), n = L%16)
__device__ __forceinline__ void store_tile(float* d, int mbase, int stride, int nbase,
                                           int lane, v8f acc, int mlimit) {
  const int roff = (lane >> 4) << 3;
  const int n = nbase + (lane & 15);
#pragma unroll
  for (int r = 0; r < 8; ++r) {
    int row = mbase + roff + r;
    if (row < mlimit) d[(size_t)row * stride + n] = acc[r];
  }
}

__global__ __launch_bounds__(256, 2)
void gat_fused_kernel(const float* __restrict__ h,      // [B,57,128]
                      const float* __restrict__ W_l,    // [256,128]
                      const float* __restrict__ w_attn, // [32]
                      const float* __restrict__ adj,    // [57,57]
                      float* __restrict__ out) {        // [B,57,256]
  extern __shared__ char smem_raw[];
  float* g_s   = reinterpret_cast<float*>(smem_raw);                         // 64 x 257 f32
  float* adj_s = reinterpret_cast<float*>(smem_raw + (size_t)NP * GS_STRIDE * sizeof(float));
  float* e_s   = adj_s + (size_t)NP * ES_STRIDE;                             // 64 x 65 f32
  float* e2_s  = e_s   + (size_t)NP * ES_STRIDE;                             // 64 x 65 f32
  __shared__ float wattn_s[F_HID];

  const int b    = blockIdx.x;
  const int tid  = threadIdx.x;
  const int lane = tid & 31;
  const int wave = tid >> 5;

  const float* hb = h + (size_t)b * N_NODES * F_IN;
  __builtin_prefetch(W_l, 0, 0);  // global_prefetch_b8: pull weights toward L2

  if (tid < F_HID) wattn_s[tid] = w_attn[tid];

  // ---- Stage adj into zero-padded LDS, overlapped with the GEMM below.
  // Zero only the padding cells (disjoint from the async-copied body).
  for (int idx = tid; idx < NP * ES_STRIDE; idx += 256) {
    int r = idx / ES_STRIDE, c = idx - r * ES_STRIDE;
    if (r >= N_NODES || c >= N_NODES) adj_s[idx] = 0.0f;
  }
#if HAVE_ASYNC_LDS
  for (int idx = tid; idx < N_NODES * N_NODES; idx += 256) {
    int r = idx / N_NODES, c = idx - r * N_NODES;
    __builtin_amdgcn_global_load_async_to_lds_b32(
        (ga_int*)(adj + idx),
        (ls_int*)(adj_s + (size_t)r * ES_STRIDE + c),
        0, 0);
  }
#else
  for (int idx = tid; idx < N_NODES * N_NODES; idx += 256) {
    int r = idx / N_NODES, c = idx - r * N_NODES;
    adj_s[(size_t)r * ES_STRIDE + c] = adj[idx];
  }
#endif

  // ---- GEMM: g = h[b] @ W_l^T (64x128 @ 128x256), operands straight from
  // global (L2-hot); tiles 4(m) x 16(n) over 8 waves; K = 4 steps of 32.
  {
    const int mt  = wave & 3;
    const int nt0 = (wave >> 2) * 8;
    v16h afr[4];
#pragma unroll
    for (int ks = 0; ks < 4; ++ks)
      afr[ks] = a_frag_global(hb, mt * 16, ks * 32, lane);
    for (int nti = 0; nti < 8; ++nti) {
      const int nt = nt0 + nti;
      v8f acc = {0.f, 0.f, 0.f, 0.f, 0.f, 0.f, 0.f, 0.f};
#pragma unroll
      for (int ks = 0; ks < 4; ++ks) {
        v16h bfr = b_fragT_global(W_l, nt * 16, ks * 32, lane);
        acc = WMMA_F16(afr[ks], bfr, acc);
      }
      store_tile(g_s, mt * 16, GS_STRIDE, nt * 16, lane, acc, NP);
    }
  }

#if HAVE_ASYNC_LDS
# if HAVE_WAIT_ASYNC_BUILTIN
  __builtin_amdgcn_s_wait_asynccnt(0);
# else
  asm volatile("s_wait_asynccnt 0" ::: "memory");
# endif
#endif
  __syncthreads();

  // ---- Per-head attention, fully in LDS
  for (int hh = 0; hh < N_HEADS; ++hh) {
    const float* gh = g_s + hh * F_HID;  // g[:, hh*32 : hh*32+32], stride GS_STRIDE

    // (a) e[i][k] = sum_f LeakyReLU(g_i[f] + g_k[f]) * w[f]   (VALU; not a matmul)
    {
      const int i  = tid & 63;
      const int k0 = (tid >> 6) << 4;  // uniform per wave -> g_k reads broadcast
      float gi[F_HID];
#pragma unroll
      for (int f = 0; f < F_HID; ++f) gi[f] = gh[(size_t)i * GS_STRIDE + f];
      for (int kk = 0; kk < 16; ++kk) {
        const int k = k0 + kk;
        float acc = 0.0f;
#pragma unroll
        for (int f = 0; f < F_HID; ++f) {
          float x = gi[f] + gh[(size_t)k * GS_STRIDE + f];
          x = (x >= 0.0f) ? x : 0.2f * x;           // LeakyReLU(alpha=0.2)
          acc = fmaf(x, wattn_s[f], acc);
        }
        e_s[(size_t)i * ES_STRIDE + k] = acc;
      }
    }
    __syncthreads();

    // (b) e2[i][j] = sum_k e[i][k] * adj[j][k]  -> e2 = e @ adj^T  (WMMA 64x64x64)
    {
      const int mt  = wave >> 1;        // 4 m-tiles
      const int ntb = (wave & 1) * 2;   // 2 n-tiles per wave
      v16h afr[2];
#pragma unroll
      for (int ks = 0; ks < 2; ++ks)
        afr[ks] = a_frag_f32(e_s, mt * 16, ES_STRIDE, ks * 32, lane);
#pragma unroll
      for (int nti = 0; nti < 2; ++nti) {
        const int nt = ntb + nti;
        v8f acc = {0.f, 0.f, 0.f, 0.f, 0.f, 0.f, 0.f, 0.f};
#pragma unroll
        for (int ks = 0; ks < 2; ++ks) {
          v16h bfr = b_fragT_f32(adj_s, nt * 16, ES_STRIDE, ks * 32, lane);
          acc = WMMA_F16(afr[ks], bfr, acc);
        }
        store_tile(e2_s, mt * 16, ES_STRIDE, nt * 16, lane, acc, NP);
      }
    }
    __syncthreads();

    // (c) softmax over j (axis=2), rows of e2; zero padded cols 57..63
    if (tid < NP) {
      float* row = e2_s + (size_t)tid * ES_STRIDE;
      float m = -1e30f;
      for (int j = 0; j < N_NODES; ++j) m = fmaxf(m, row[j]);
      float s = 0.0f;
      for (int j = 0; j < N_NODES; ++j) { float v = __expf(row[j] - m); row[j] = v; s += v; }
      float inv = 1.0f / s;
      for (int j = 0; j < N_NODES; ++j) row[j] *= inv;
      for (int j = N_NODES; j < NP; ++j) row[j] = 0.0f;
    }
    __syncthreads();

    // (d) attn_res[i][f] = sum_j a[i][j] * g[j][hh*32+f]  (WMMA, 64x64 @ 64x32)
    {
      const int mt = wave & 3;
      const int nt = wave >> 2;  // 2 n-tiles cover F_HID=32
      v8f acc = {0.f, 0.f, 0.f, 0.f, 0.f, 0.f, 0.f, 0.f};
#pragma unroll
      for (int ks = 0; ks < 2; ++ks) {
        v16h afr = a_frag_f32(e2_s, mt * 16, ES_STRIDE, ks * 32, lane);
        v16h bfr = b_fragN_f32(gh, ks * 32, GS_STRIDE, nt * 16, lane);
        acc = WMMA_F16(afr, bfr, acc);
      }
      const int roff = (lane >> 4) << 3;
      const int n = nt * 16 + (lane & 15);
#pragma unroll
      for (int r = 0; r < 8; ++r) {
        int row = mt * 16 + roff + r;
        if (row < N_NODES)
          out[(size_t)b * N_NODES * F_OUT + (size_t)row * F_OUT + hh * F_HID + n] = acc[r];
      }
    }
    __syncthreads();  // protect e_s/e2_s reuse next head
  }
}

extern "C" void kernel_launch(void* const* d_in, const int* in_sizes, int n_in,
                              void* d_out, int out_size, void* d_ws, size_t ws_size,
                              hipStream_t stream) {
  (void)n_in; (void)out_size; (void)d_ws; (void)ws_size;
  const float* h      = (const float*)d_in[0];
  const float* W_l    = (const float*)d_in[1];
  const float* w_attn = (const float*)d_in[2];
  const float* adj    = (const float*)d_in[3];
  float* out          = (float*)d_out;

  const int B = in_sizes[0] / (N_NODES * F_IN);  // 256

  // dynamic LDS: g (64*257 f32) + adj/e/e2 (3x 64*65 f32) = 115,712 B
  const size_t smem = (size_t)NP * GS_STRIDE * sizeof(float)
                    + 3u * (size_t)NP * ES_STRIDE * sizeof(float);

  gat_fused_kernel<<<dim3(B), dim3(256), smem, stream>>>(h, W_l, w_attn, adj, out);
}
